// MultiHeadAttention_60851096649901
// MI455X (gfx1250) — matrix operand order repaired and checked by reading, not compiled, not run on hardware
//
#include <hip/hip_runtime.h>
#include <stdint.h>

// ---------------------------------------------------------------------------
// MI455X (gfx1250) multi-head attention forward.
// - All GEMMs via v_wmma_f32_16x16x32_bf16 (wave32 WMMA, f32 accumulate)
// - GEMM A-tiles staged with global_load_async_to_lds_b128 (ASYNCcnt)
// - Attention Q/K tiles staged with TDM tensor_load_to_lds (TENSORcnt)
// ---------------------------------------------------------------------------

typedef unsigned short u16;
typedef __bf16    v16bf __attribute__((ext_vector_type(16)));
typedef float     v8f   __attribute__((ext_vector_type(8)));
typedef uint32_t  v8u   __attribute__((ext_vector_type(8)));
typedef uint32_t  u32x4 __attribute__((ext_vector_type(4)));
typedef int       i32x8 __attribute__((ext_vector_type(8)));
typedef int       i32x4 __attribute__((ext_vector_type(4)));

#if __has_builtin(__builtin_amdgcn_tensor_load_to_lds)
#define HAVE_TDM 1
#else
#define HAVE_TDM 0
#endif

__device__ __forceinline__ u16 f2bf(float f) {
    uint32_t u = __float_as_uint(f);
    u += 0x7FFFu + ((u >> 16) & 1u);          // round-to-nearest-even
    return (u16)(u >> 16);
}

__device__ __forceinline__ v16bf pack_frag(uint4 a, uint4 b) {
    v8u r;
    r[0] = a.x; r[1] = a.y; r[2] = a.z; r[3] = a.w;
    r[4] = b.x; r[5] = b.y; r[6] = b.z; r[7] = b.w;
    return __builtin_bit_cast(v16bf, r);
}

__device__ __forceinline__ v8f wmma_bf16(v16bf a, v16bf b, v8f c) {
    return __builtin_amdgcn_wmma_f32_16x16x32_bf16(
        false, a, false, b, (short)0, c, false, false);
}

// Generic pointer into __shared__ -> LDS byte address (LDS aperture maps the
// low 32 bits of the generic address directly onto the LDS offset).
__device__ __forceinline__ uint32_t lds_addr_of(const void* p) {
    return (uint32_t)(uintptr_t)p;
}

// Per-lane async global->LDS 16-byte copy (tracked with ASYNCcnt).
__device__ __forceinline__ void async_load_b128(uint32_t lds_off, const void* gaddr) {
    asm volatile("global_load_async_to_lds_b128 %0, %1, off"
                 :: "v"(lds_off), "v"(gaddr) : "memory");
}
__device__ __forceinline__ void wait_asynccnt0() {
    asm volatile("s_wait_asynccnt 0x0" ::: "memory");
}

__device__ __forceinline__ void wait_tensorcnt0() {
#if __has_builtin(__builtin_amdgcn_s_wait_tensorcnt)
    __builtin_amdgcn_s_wait_tensorcnt(0);
#else
    asm volatile("s_wait_tensorcnt 0x0" ::: "memory");
#endif
}

#if HAVE_TDM
// TDM 2D tile load: bf16 elements (data_size=2B).  Tile (tile_d0 x tile_d1)
// from a row-major tensor with dim0-stride `stride_elems`, into LDS at
// `lds_addr` (packed, tile_d0 contiguous per row).  Issued once per wave.
// NOTE: this toolchain exposes the 6-arg builtin form:
//   (u32x4 group0, i32x8 group1, i32x4 group2, i32x4 group3, i32x8, i32 cpol)
__device__ __forceinline__ void tdm_load_tile_2d(
    uint32_t lds_addr, const void* gptr,
    uint32_t tile_d0, uint32_t tile_d1,
    uint32_t tensor_d0, uint32_t tensor_d1, uint32_t stride_elems)
{
    const uint64_t ga = (uint64_t)(uintptr_t)gptr;
    u32x4 g0;
    g0[0] = 1u;                                   // count=1, user descriptor
    g0[1] = lds_addr;                             // lds_addr [63:32]
    g0[2] = (uint32_t)ga;                         // global_addr low
    g0[3] = (uint32_t)(ga >> 32) | (2u << 30);    // global_addr high | type=2
    i32x8 g1;
    g1[0] = (int)(1u << 16);                      // wg_mask=0, data_size=1 (2B)
    g1[1] = (int)(tensor_d0 << 16);               // barrier_addr=0 | tensor_dim0.lo
    g1[2] = (int)((tensor_d0 >> 16) | (tensor_d1 << 16));
    g1[3] = (int)((tensor_d1 >> 16) | (tile_d0 << 16));
    g1[4] = (int)tile_d1;                         // tile_dim1 | tile_dim2=0
    g1[5] = (int)stride_elems;                    // tensor_dim0_stride.lo
    g1[6] = 0;
    g1[7] = 0;                                    // dim1 stride unused (2D)
    i32x4 z4; z4[0] = 0; z4[1] = 0; z4[2] = 0; z4[3] = 0;
    i32x8 z8;
#pragma unroll
    for (int i = 0; i < 8; ++i) z8[i] = 0;
    __builtin_amdgcn_tensor_load_to_lds(g0, g1, z4, z4, z8, 0);
}
#endif

// ---------------------------------------------------------------------------
// fp32 -> bf16 conversion (vectorized, n divisible by 4)
// ---------------------------------------------------------------------------
__global__ __launch_bounds__(256) void cvt_bf16_kernel(
    const float* __restrict__ in, u16* __restrict__ out, int n4)
{
    int i = blockIdx.x * 256 + threadIdx.x;
    if (i >= n4) return;
    float4 f = reinterpret_cast<const float4*>(in)[i];
    uint2 o;
    o.x = (uint32_t)f2bf(f.x) | ((uint32_t)f2bf(f.y) << 16);
    o.y = (uint32_t)f2bf(f.z) | ((uint32_t)f2bf(f.w) << 16);
    reinterpret_cast<uint2*>(out)[i] = o;
}

// ---------------------------------------------------------------------------
// C[m,n] = A[m,:] @ W[:,n] + bias[n].   A: [M,1024] bf16 RM, W: [1024,1024]
// bf16 RM, bias fp32.  128x128 block tile, 8 waves (each: 16 M x 128 N),
// K-step 32.  A staged via async global->LDS; B staged transposed manually.
// ---------------------------------------------------------------------------
template <bool OUT_F32>
__global__ __launch_bounds__(256) void gemm_bf16_wmma(
    const u16* __restrict__ A, const u16* __restrict__ W,
    const float* __restrict__ bias, u16* __restrict__ Cb,
    float* __restrict__ Cf)
{
    constexpr int K = 1024, N = 1024;
    __shared__ __align__(16) u16 Alds[128 * 32];   // [m][k]
    __shared__ __align__(16) u16 Blds[128 * 32];   // [n][k] (transposed)

    const int tid  = threadIdx.x;
    const int wave = tid >> 5, lane = tid & 31;
    const int lrow = lane & 15, half = lane >> 4;
    const int M0 = blockIdx.x * 128, N0 = blockIdx.y * 128;

    v8f acc[8];
#pragma unroll
    for (int i = 0; i < 8; ++i)
#pragma unroll
        for (int e = 0; e < 8; ++e) acc[i][e] = 0.0f;

    const int bn  = tid & 127;  // n within tile for B-transpose staging
    const int bkg = tid >> 7;   // 0..1 -> k-local 16*bkg..+15

    for (int k0 = 0; k0 < K; k0 += 32) {
        __syncthreads();
        // ---- stage A tile (128x32): async global->LDS b128, 2 per lane ----
#pragma unroll
        for (int i = 0; i < 2; ++i) {
            int u = tid + i * 256;          // uint4 units (512 total)
            int row = u >> 2, ku = u & 3;
            async_load_b128(lds_addr_of(&Alds[row * 32 + ku * 8]),
                            A + (size_t)(M0 + row) * K + k0 + ku * 8);
        }
        // ---- stage B tile transposed: thread owns column n, 16 k values ----
        {
            uint32_t w[8];
#pragma unroll
            for (int j = 0; j < 16; ++j) {
                u16 t = W[(size_t)(k0 + bkg * 16 + j) * N + N0 + bn];
                if ((j & 1) == 0) w[j >> 1] = t;
                else              w[j >> 1] |= (uint32_t)t << 16;
            }
            *reinterpret_cast<uint4*>(&Blds[bn * 32 + bkg * 16])     = make_uint4(w[0], w[1], w[2], w[3]);
            *reinterpret_cast<uint4*>(&Blds[bn * 32 + bkg * 16 + 8]) = make_uint4(w[4], w[5], w[6], w[7]);
            if (k0 + 32 < K)
                __builtin_prefetch(W + (size_t)(k0 + 32 + bkg * 16) * N + N0 + bn, 0, 0);
        }
        wait_asynccnt0();
        __syncthreads();

        // ---- A fragment (16x32): VGPR0-3 = K(8*half)..+7, VGPR4-7 = K(16+8*half)..+7
        const int arow = wave * 16 + lrow;
        uint4 a0 = *reinterpret_cast<const uint4*>(&Alds[arow * 32 + 8 * half]);
        uint4 a1 = *reinterpret_cast<const uint4*>(&Alds[arow * 32 + 16 + 8 * half]);
        v16bf af = pack_frag(a0, a1);
#pragma unroll
        for (int nc = 0; nc < 8; ++nc) {
            const int brow = nc * 16 + lrow;
            uint4 b0 = *reinterpret_cast<const uint4*>(&Blds[brow * 32 + 8 * half]);
            uint4 b1 = *reinterpret_cast<const uint4*>(&Blds[brow * 32 + 16 + 8 * half]);
            acc[nc] = wmma_bf16(af, pack_frag(b0, b1), acc[nc]);
        }
    }

    // ---- epilogue: bias + store ----
#pragma unroll
    for (int nc = 0; nc < 8; ++nc) {
        const int n = N0 + nc * 16 + lrow;
        const float bv = bias[n];
#pragma unroll
        for (int r = 0; r < 8; ++r) {
            const int m = M0 + wave * 16 + r + 8 * half;
            const float v = acc[nc][r] + bv;
            if constexpr (OUT_F32) Cf[(size_t)m * N + n] = v;
            else                   Cb[(size_t)m * N + n] = f2bf(v);
        }
    }
}

// ---------------------------------------------------------------------------
// Flash attention.  Q/K/V: bf16 [B=4][S=2048][H=16][D=64].
// Block = 256 threads = 8 waves; one (b,h), 128 q-rows (16/wave).
// K-tile = 64.  scores = (Q K^T) * sqrt(D) = * 8.0, online softmax, P@V.
// Q and K tiles staged via TDM tensor_load_to_lds; V transposed manually.
// ---------------------------------------------------------------------------
__global__ __launch_bounds__(256) void attn_flash_wmma(
    const u16* __restrict__ Q, const u16* __restrict__ Kmat,
    const u16* __restrict__ V, u16* __restrict__ Ctx)
{
    constexpr int S = 2048, H = 16, D = 64, SSTR = H * D; // 1024
    __shared__ __align__(16) u16 Qlds[128 * 64];   // [q][d]
    __shared__ __align__(16) u16 Klds[64 * 64];    // [kpos][d]  == B^T layout for QK^T
    __shared__ __align__(16) u16 VT[64 * 64];      // [d][kpos]  == B^T layout for P@V
    __shared__ __align__(16) u16 Plds[8 * 16 * 64];// per-wave P [q16][k64]

    const int tid  = threadIdx.x;
    const int wave = tid >> 5, lane = tid & 31;
    const int lrow = lane & 15, half = lane >> 4;
    const int b = blockIdx.y >> 4, h = blockIdx.y & 15;
    const int q0 = blockIdx.x * 128;
    const size_t base = ((size_t)b * S * H + h) * D;

    // ---- stage Q once: 128x64 tile, stride 1024 elems ----
#if HAVE_TDM
    if (wave == 0)
        tdm_load_tile_2d(lds_addr_of(&Qlds[0]), Q + base + (size_t)q0 * SSTR,
                         /*tile*/ D, 128, /*tensor*/ D, S, SSTR);
#else
#pragma unroll
    for (int i = 0; i < 4; ++i) {
        int u = tid + i * 256;
        int row = u >> 3, du = u & 7;
        const uint4* src = reinterpret_cast<const uint4*>(
            Q + base + (size_t)(q0 + row) * SSTR + du * 8);
        *reinterpret_cast<uint4*>(&Qlds[row * 64 + du * 8]) = *src;
    }
#endif

    float mi[8], li[8];
#pragma unroll
    for (int r = 0; r < 8; ++r) { mi[r] = -3.0e38f; li[r] = 0.0f; }
    v8f oacc[4];
#pragma unroll
    for (int i = 0; i < 4; ++i)
#pragma unroll
        for (int e = 0; e < 8; ++e) oacc[i][e] = 0.0f;

    u16* myP = &Plds[wave * 16 * 64];
    const int qrow = wave * 16 + lrow;

    for (int kt = 0; kt < S; kt += 64) {
        __syncthreads();
        // ---- stage K tile (64x64 row-major) via TDM ----
#if HAVE_TDM
        if (wave == 0)
            tdm_load_tile_2d(lds_addr_of(&Klds[0]), Kmat + base + (size_t)kt * SSTR,
                             /*tile*/ D, 64, /*tensor*/ D, S, SSTR);
#else
#pragma unroll
        for (int i = 0; i < 2; ++i) {
            int u = tid + i * 256;
            int row = u >> 3, du = u & 7;
            const uint4* src = reinterpret_cast<const uint4*>(
                Kmat + base + (size_t)(kt + row) * SSTR + du * 8);
            *reinterpret_cast<uint4*>(&Klds[row * 64 + du * 8]) = *src;
        }
#endif
        // ---- stage V tile transposed: thread owns d, gathers 16 k ----
        {
            int d = tid & 63, kg = tid >> 6;   // kg = 0..3
            uint32_t w[8];
#pragma unroll
            for (int j = 0; j < 16; ++j) {
                u16 t = V[base + (size_t)(kt + kg * 16 + j) * SSTR + d];
                if ((j & 1) == 0) w[j >> 1] = t;
                else              w[j >> 1] |= (uint32_t)t << 16;
            }
            *reinterpret_cast<uint4*>(&VT[d * 64 + kg * 16])     = make_uint4(w[0], w[1], w[2], w[3]);
            *reinterpret_cast<uint4*>(&VT[d * 64 + kg * 16 + 8]) = make_uint4(w[4], w[5], w[6], w[7]);
        }
#if HAVE_TDM
        if (wave == 0) wait_tensorcnt0();   // covers Q (first iter) + this K
#endif
        __syncthreads();

        // ---- scores S = Q K^T  (16 q x 64 k per wave), K-dim = D = 64 ----
        v8f sacc[4];
#pragma unroll
        for (int i = 0; i < 4; ++i)
#pragma unroll
            for (int e = 0; e < 8; ++e) sacc[i][e] = 0.0f;
#pragma unroll
        for (int ks = 0; ks < 64; ks += 32) {
            uint4 a0 = *reinterpret_cast<const uint4*>(&Qlds[qrow * 64 + ks + 8 * half]);
            uint4 a1 = *reinterpret_cast<const uint4*>(&Qlds[qrow * 64 + ks + 16 + 8 * half]);
            v16bf af = pack_frag(a0, a1);
#pragma unroll
            for (int nc = 0; nc < 4; ++nc) {
                const int kr = nc * 16 + lrow;
                uint4 b0 = *reinterpret_cast<const uint4*>(&Klds[kr * 64 + ks + 8 * half]);
                uint4 b1 = *reinterpret_cast<const uint4*>(&Klds[kr * 64 + ks + 16 + 8 * half]);
                sacc[nc] = wmma_bf16(af, pack_frag(b0, b1), sacc[nc]);
            }
        }

        // ---- scale by sqrt(D)=8, online softmax over this 64-wide slab ----
#pragma unroll
        for (int r = 0; r < 8; ++r) {
            float mrow = -3.0e38f;
#pragma unroll
            for (int nc = 0; nc < 4; ++nc) {
                sacc[nc][r] *= 8.0f;
                mrow = fmaxf(mrow, sacc[nc][r]);
            }
#pragma unroll
            for (int msk = 1; msk < 16; msk <<= 1)
                mrow = fmaxf(mrow, __shfl_xor(mrow, msk, 32));
            const float mnew = fmaxf(mi[r], mrow);
            const float corr = __expf(mi[r] - mnew);
            float psum = 0.0f;
#pragma unroll
            for (int nc = 0; nc < 4; ++nc) {
                float p = __expf(sacc[nc][r] - mnew);
                sacc[nc][r] = p;
                psum += p;
            }
#pragma unroll
            for (int msk = 1; msk < 16; msk <<= 1)
                psum += __shfl_xor(psum, msk, 32);
            li[r] = li[r] * corr + psum;
            mi[r] = mnew;
#pragma unroll
            for (int nc = 0; nc < 4; ++nc) oacc[nc][r] *= corr;
        }

        // ---- re-layout P through per-wave LDS (C-layout -> A-fragment) ----
#pragma unroll
        for (int nc = 0; nc < 4; ++nc)
#pragma unroll
            for (int r = 0; r < 8; ++r)
                myP[(r + 8 * half) * 64 + nc * 16 + lrow] = f2bf(sacc[nc][r]);

        // ---- O += P @ V  (K-dim = 64 k-positions) ----
#pragma unroll
        for (int ks = 0; ks < 64; ks += 32) {
            uint4 a0 = *reinterpret_cast<const uint4*>(&myP[lrow * 64 + ks + 8 * half]);
            uint4 a1 = *reinterpret_cast<const uint4*>(&myP[lrow * 64 + ks + 16 + 8 * half]);
            v16bf af = pack_frag(a0, a1);
#pragma unroll
            for (int nc = 0; nc < 4; ++nc) {
                const int dr = nc * 16 + lrow;
                uint4 b0 = *reinterpret_cast<const uint4*>(&VT[dr * 64 + ks + 8 * half]);
                uint4 b1 = *reinterpret_cast<const uint4*>(&VT[dr * 64 + ks + 16 + 8 * half]);
                oacc[nc] = wmma_bf16(af, pack_frag(b0, b1), oacc[nc]);
            }
        }
    }

    // ---- epilogue: normalize, store bf16 context [B,S,H,D] ----
#pragma unroll
    for (int nc = 0; nc < 4; ++nc) {
        const int d = nc * 16 + lrow;
#pragma unroll
        for (int r = 0; r < 8; ++r) {
            const int qg = q0 + wave * 16 + r + 8 * half;
            const float v = oacc[nc][r] / li[r];
            Ctx[base + (size_t)qg * SSTR + d] = f2bf(v);
        }
    }
}

// ---------------------------------------------------------------------------
// Host-side launch
// ---------------------------------------------------------------------------
extern "C" void kernel_launch(void* const* d_in, const int* in_sizes, int n_in,
                              void* d_out, int out_size, void* d_ws, size_t ws_size,
                              hipStream_t stream) {
    (void)in_sizes; (void)n_in; (void)out_size; (void)ws_size;
    constexpr int Bsz = 4, S = 2048, E = 1024;
    constexpr size_t MS = (size_t)Bsz * S;            // 8192 rows

    const float* x  = (const float*)d_in[0];
    const float* Wq = (const float*)d_in[1];
    const float* bq = (const float*)d_in[2];
    const float* Wk = (const float*)d_in[3];
    const float* bk = (const float*)d_in[4];
    const float* Wv = (const float*)d_in[5];
    const float* bv = (const float*)d_in[6];
    const float* Wo = (const float*)d_in[7];
    const float* bo = (const float*)d_in[8];
    float* out = (float*)d_out;

    uint8_t* ws = (uint8_t*)d_ws;
    u16* xb  = (u16*)(ws);                                   // 8192x1024
    u16* Wqb = (u16*)(ws + 16777216);                        // 1024x1024
    u16* Wkb = (u16*)(ws + 18874368);
    u16* Wvb = (u16*)(ws + 20971520);
    u16* Wob = (u16*)(ws + 23068672);
    u16* Qb  = (u16*)(ws + 25165824);                        // 8192x1024
    u16* Kb  = (u16*)(ws + 41943040);
    u16* Vb  = (u16*)(ws + 58720256);
    u16* Ctx = (u16*)(ws + 75497472);

    // fp32 -> bf16
    {
        int n4 = (int)(MS * E / 4);
        cvt_bf16_kernel<<<(n4 + 255) / 256, 256, 0, stream>>>(x, xb, n4);
        int w4 = E * E / 4;
        cvt_bf16_kernel<<<(w4 + 255) / 256, 256, 0, stream>>>(Wq, Wqb, w4);
        cvt_bf16_kernel<<<(w4 + 255) / 256, 256, 0, stream>>>(Wk, Wkb, w4);
        cvt_bf16_kernel<<<(w4 + 255) / 256, 256, 0, stream>>>(Wv, Wvb, w4);
        cvt_bf16_kernel<<<(w4 + 255) / 256, 256, 0, stream>>>(Wo, Wob, w4);
    }

    // Q/K/V projections (bf16 out, bias fused)
    dim3 gg(64, 8);           // 8192/128 x 1024/128
    gemm_bf16_wmma<false><<<gg, 256, 0, stream>>>(xb, Wqb, bq, Qb, nullptr);
    gemm_bf16_wmma<false><<<gg, 256, 0, stream>>>(xb, Wkb, bk, Kb, nullptr);
    gemm_bf16_wmma<false><<<gg, 256, 0, stream>>>(xb, Wvb, bv, Vb, nullptr);

    // flash attention: grid = (S/128 q-tiles, B*H)
    attn_flash_wmma<<<dim3(16, 64), 256, 0, stream>>>(Qb, Kb, Vb, Ctx);

    // output projection (fp32 out)
    gemm_bf16_wmma<true><<<gg, 256, 0, stream>>>(Ctx, Wob, bo, nullptr, out);
}
